// Block_67697274519925
// MI455X (gfx1250) — matrix operand order repaired and checked by reading, not compiled, not run on hardware
//
#include <hip/hip_runtime.h>
#include <hip/hip_bf16.h>
#include <math.h>

typedef __attribute__((ext_vector_type(16))) _Float16 v16h;
typedef __attribute__((ext_vector_type(8)))  _Float16 v8h;
typedef __attribute__((ext_vector_type(8)))  float    v8f;

#define BN_   4
#define HS_   56
#define WS_   56
#define NPIX  (HS_*WS_)      // 3136
#define DIM_  256
#define HEADS_ 8
#define HD_   32
#define HK_   (HS_/2)
#define WK_   (WS_/2)
#define NK_   (HK_*WK_)      // 784
#define HID_  1024
#define EPS_  1e-5f

__device__ __forceinline__ float gelu_exact(float x) {
  return 0.5f * x * (1.0f + erff(x * 0.7071067811865475f));
}
__device__ __forceinline__ float bn_apply(float x, float g, float b, float m, float v) {
  return (x - m) * (g * rsqrtf(v + EPS_)) + b;
}

// ---------------- f32 -> f16 convert ----------------
__global__ __launch_bounds__(256) void k_cvt(const float* __restrict__ s,
                                             _Float16* __restrict__ d, int n) {
  int i = blockIdx.x * 256 + threadIdx.x;
  if (i < n) d[i] = (_Float16)s[i];
}

// ---------------- LPU: depthwise 3x3 + residual ----------------
__global__ __launch_bounds__(256) void k_lpu(const float* __restrict__ x,
                                             const float* __restrict__ w,
                                             const float* __restrict__ bias,
                                             float* __restrict__ y) {
  int bn = blockIdx.x;                  // b*NPIX + n
  int c  = threadIdx.x;
  int b  = bn / NPIX, n = bn % NPIX;
  int h  = n / WS_, wc = n % WS_;
  float acc = bias[c];
#pragma unroll
  for (int dh = 0; dh < 3; ++dh)
#pragma unroll
    for (int dw = 0; dw < 3; ++dw) {
      int hh = h + dh - 1, ww = wc + dw - 1;
      if (hh >= 0 && hh < HS_ && ww >= 0 && ww < WS_)
        acc += x[((size_t)b * NPIX + hh * WS_ + ww) * DIM_ + c] * w[c * 9 + dh * 3 + dw];
    }
  size_t idx = (size_t)bn * DIM_ + c;
  y[idx] = acc + x[idx];
}

// ---------------- LayerNorm over C=256, f16 out ----------------
__global__ __launch_bounds__(256) void k_ln(const float* __restrict__ x,
                                            const float* __restrict__ g,
                                            const float* __restrict__ b,
                                            _Float16* __restrict__ y) {
  __shared__ float ss[256], sq[256];
  int row = blockIdx.x, t = threadIdx.x;
  float v = x[(size_t)row * DIM_ + t];
  ss[t] = v; sq[t] = v * v;
  __syncthreads();
  for (int s = 128; s > 0; s >>= 1) {
    if (t < s) { ss[t] += ss[t + s]; sq[t] += sq[t + s]; }
    __syncthreads();
  }
  float mu  = ss[0] * (1.0f / DIM_);
  float var = sq[0] * (1.0f / DIM_) - mu * mu;
  float r   = rsqrtf(var + EPS_);
  y[(size_t)row * DIM_ + t] = (_Float16)((v - mu) * r * g[t] + b[t]);
}

// ---------------- SR: depthwise 2x2 stride-2 + BN, f16 out ----------------
__global__ __launch_bounds__(256) void k_sr(const _Float16* __restrict__ xn,
                                            const float* __restrict__ w,
                                            const float* __restrict__ bias,
                                            const float* __restrict__ g,
                                            const float* __restrict__ bb,
                                            const float* __restrict__ mm,
                                            const float* __restrict__ vv,
                                            _Float16* __restrict__ xs) {
  int bn = blockIdx.x;                  // b*NK_ + nk
  int c  = threadIdx.x;
  int b  = bn / NK_, nk = bn % NK_;
  int hs = nk / WK_, ws = nk % WK_;
  float acc = bias[c];
#pragma unroll
  for (int dh = 0; dh < 2; ++dh)
#pragma unroll
    for (int dw = 0; dw < 2; ++dw) {
      int hh = hs * 2 + dh, ww = ws * 2 + dw;
      acc += (float)xn[((size_t)b * NPIX + hh * WS_ + ww) * DIM_ + c] * w[c * 4 + dh * 2 + dw];
    }
  acc = bn_apply(acc, g[c], bb[c], mm[c], vv[c]);
  xs[(size_t)bn * DIM_ + c] = (_Float16)acc;
}

// ---------------- generic WMMA GEMM: C[M,N] = A[M,K] * W[N,K]^T ----------------
// Double-buffered GLOBAL_LOAD_ASYNC_TO_LDS pipeline for the A tile (16x64 per
// stage); the async DMA for stage s+1 is in flight while stage s feeds 2 WMMAs.
// modes: 0 plain f16 store; 1 f16 transposed store (per-batch [C,Nk]);
//        2 +bias +resid -> f32; 3 +bias gelu bn -> f16; 4 +bias bn +resid -> f32
__global__ __launch_bounds__(256) void k_gemm(const _Float16* __restrict__ A,
                                              const _Float16* __restrict__ Bw,
                                              int M, int Nt, int K, int mode,
                                              const float* __restrict__ bias,
                                              const float* __restrict__ bn_g,
                                              const float* __restrict__ bn_b,
                                              const float* __restrict__ bn_m,
                                              const float* __restrict__ bn_v,
                                              const float* __restrict__ resid,
                                              float* __restrict__ outF,
                                              _Float16* __restrict__ outH,
                                              int tdim) {
  __shared__ __align__(32) _Float16 Atile[2][16 * 64];
  const int t    = threadIdx.x;
  const int wave = t >> 5, lane = t & 31;
  const int hi   = lane >> 4, lm = lane & 15;
  const int M0   = blockIdx.x * 16;
  const int nCol = blockIdx.y * 128 + wave * 16 + lm;
  const int idx  = t * 4;               // 4 halves (8B) per thread per stage
  const int ar   = idx >> 6, ac = idx & 63;
  const _Float16* aRow = A + (size_t)(M0 + ar) * K + ac;

  v8f acc = {};
  const int nStages = K >> 6;

  // prologue: async DMA stage 0 -> buffer 0
  {
    unsigned ldsa = (unsigned)(uintptr_t)&Atile[0][idx];
    unsigned long long ga = (unsigned long long)aRow;
    asm volatile("global_load_async_to_lds_b64 %0, %1, off"
                 :: "v"(ldsa), "v"(ga) : "memory");
  }
  for (int s = 0; s < nStages; ++s) {
    __syncthreads();  // everyone finished reading buffer (s+1)&1 in stage s-1
    if (s + 1 < nStages) {
      unsigned ldsa = (unsigned)(uintptr_t)&Atile[(s + 1) & 1][idx];
      unsigned long long ga = (unsigned long long)(aRow + (s + 1) * 64);
      asm volatile("global_load_async_to_lds_b64 %0, %1, off"
                   :: "v"(ldsa), "v"(ga) : "memory");
      // oldest outstanding copy (stage s) is done; next stays in flight
      asm volatile("s_wait_asynccnt 0x1" ::: "memory");
      __builtin_prefetch(Bw + (size_t)nCol * K + (s + 1) * 64 + hi * 16, 0, 3);
    } else {
      asm volatile("s_wait_asynccnt 0x0" ::: "memory");
    }
    __syncthreads();
    const _Float16* At = &Atile[s & 1][0];
#pragma unroll
    for (int sub = 0; sub < 64; sub += 32) {
      union { v16h v; v8h h2[2]; } af, bf;
      af.h2[0] = *reinterpret_cast<const v8h*>(&At[lm * 64 + sub + hi * 16]);
      af.h2[1] = *reinterpret_cast<const v8h*>(&At[lm * 64 + sub + hi * 16 + 8]);
      const _Float16* bp = Bw + (size_t)nCol * K + s * 64 + sub + hi * 16;
      bf.v = *reinterpret_cast<const v16h*>(bp);
      acc = __builtin_amdgcn_wmma_f32_16x16x32_f16(false, af.v, false, bf.v,
                                                   (short)0, acc, false, false);
    }
  }
#pragma unroll
  for (int r = 0; r < 8; ++r) {
    int row = M0 + hi * 8 + r;
    int n   = nCol;
    float c = acc[r];
    if (mode == 0) {
      outH[(size_t)row * Nt + n] = (_Float16)c;
    } else if (mode == 1) {
      int bb2 = row / tdim, nk = row % tdim;
      outH[((size_t)bb2 * Nt + n) * tdim + nk] = (_Float16)c;
    } else if (mode == 2) {
      outF[(size_t)row * Nt + n] = c + bias[n] + resid[(size_t)row * Nt + n];
    } else if (mode == 3) {
      float v = gelu_exact(c + bias[n]);
      v = bn_apply(v, bn_g[n], bn_b[n], bn_m[n], bn_v[n]);
      outH[(size_t)row * Nt + n] = (_Float16)v;
    } else {
      float v = bn_apply(c + bias[n], bn_g[n], bn_b[n], bn_m[n], bn_v[n]);
      outF[(size_t)row * Nt + n] = v + resid[(size_t)row * Nt + n];
    }
  }
}

// ---------------- fused flash attention ----------------
// one wave per (b, head, 16 q-rows); keys processed in chunks of 32
__global__ __launch_bounds__(256) void k_attn(const _Float16* __restrict__ q,
                                              const _Float16* __restrict__ kk,
                                              const _Float16* __restrict__ vT,
                                              const float* __restrict__ rel,
                                              _Float16* __restrict__ o) {
  __shared__ __align__(32) _Float16 pbuf[8 * 512];
  const int lane = threadIdx.x & 31, wave = threadIdx.x >> 5;
  const int hi = lane >> 4, lm = lane & 15;
  int gw  = blockIdx.x * 8 + wave;
  int qT  = gw % (NPIX / 16);
  int tmp = gw / (NPIX / 16);
  int h   = tmp % HEADS_;
  int b   = tmp / HEADS_;
  int q0  = qT * 16;
  int hc0 = h * HD_;
  _Float16* P = &pbuf[wave * 512];

  union U16 { v16h v; v8h h2[2]; };
  const v16h z16 = {};
  const v8f  zf  = {};

  // Q A-fragment (hd=32 == K)
  U16 qf;
  const _Float16* qbase = q + ((size_t)b * NPIX + q0 + lm) * DIM_ + hc0 + hi * 16;
  qf.h2[0] = *reinterpret_cast<const v8h*>(qbase);
  qf.h2[1] = *reinterpret_cast<const v8h*>(qbase + 8);

  v8f accLo = {}, accHi = {};
  float mrow[8], lrow[8];
#pragma unroll
  for (int r = 0; r < 8; ++r) { mrow[r] = -__builtin_inff(); lrow[r] = 0.0f; }
  const float scale = 0.17677669529663687f;  // 1/sqrt(32)

  for (int kb = 0; kb < NK_; kb += 32) {
    // K B-fragments: keys kb..kb+15 and kb+16..kb+31
    U16 ka, kbv;
    {
      const _Float16* kp = kk + ((size_t)b * NK_ + kb + lm) * DIM_ + hc0 + hi * 16;
      ka.v = *reinterpret_cast<const v16h*>(kp);
    }
    kbv.v = z16;
    if (kb + 16 < NK_) {
      const _Float16* kp = kk + ((size_t)b * NK_ + kb + 16 + lm) * DIM_ + hc0 + hi * 16;
      kbv.v = *reinterpret_cast<const v16h*>(kp);
    }
    v8f sa = __builtin_amdgcn_wmma_f32_16x16x32_f16(false, qf.v, false, ka.v,
                                                    (short)0, zf, false, false);
    v8f sb = __builtin_amdgcn_wmma_f32_16x16x32_f16(false, qf.v, false, kbv.v,
                                                    (short)0, zf, false, false);
    // online softmax (rows hi*8+r live across the 16 lanes of each half-wave)
#pragma unroll
    for (int r = 0; r < 8; ++r) {
      int m = hi * 8 + r;
      const float* rp = rel + ((size_t)h * NPIX + q0 + m) * NK_;
      int ca = kb + lm, cb = kb + 16 + lm;
      float va = sa[r] * scale + rp[ca];
      float vb = (cb < NK_) ? (sb[r] * scale + rp[cb]) : -__builtin_inff();
      float mx = fmaxf(va, vb);
      for (int off = 1; off < 16; off <<= 1) mx = fmaxf(mx, __shfl_xor(mx, off));
      float newm = fmaxf(mrow[r], mx);
      float corr = expf(mrow[r] - newm);
      float ea = expf(va - newm);
      float eb = (cb < NK_) ? expf(vb - newm) : 0.0f;
      float rs = ea + eb;
      for (int off = 1; off < 16; off <<= 1) rs += __shfl_xor(rs, off);
      lrow[r] = lrow[r] * corr + rs;
      mrow[r] = newm;
      accLo[r] *= corr;
      accHi[r] *= corr;
      P[m * 32 + lm]      = (_Float16)ea;   // same-wave LDS ops are in-order
      P[m * 32 + 16 + lm] = (_Float16)eb;
    }
    // P (16x32) back as A-fragment (C-layout -> A-layout via per-wave LDS tile)
    U16 pf;
    pf.h2[0] = *reinterpret_cast<const v8h*>(&P[lm * 32 + hi * 16]);
    pf.h2[1] = *reinterpret_cast<const v8h*>(&P[lm * 32 + hi * 16 + 8]);
    // V B-fragments from pre-transposed V^T [b][c][nk] (contiguous 32B per lane)
    U16 vlo, vhi;
    vlo.v = z16; vhi.v = z16;
    if (kb + hi * 16 < NK_) {
      const _Float16* vp  = vT + ((size_t)b * DIM_ + hc0 + lm) * NK_ + kb + hi * 16;
      const _Float16* vp2 = vT + ((size_t)b * DIM_ + hc0 + 16 + lm) * NK_ + kb + hi * 16;
      vlo.v = *reinterpret_cast<const v16h*>(vp);
      vhi.v = *reinterpret_cast<const v16h*>(vp2);
    }
    accLo = __builtin_amdgcn_wmma_f32_16x16x32_f16(false, pf.v, false, vlo.v,
                                                   (short)0, accLo, false, false);
    accHi = __builtin_amdgcn_wmma_f32_16x16x32_f16(false, pf.v, false, vhi.v,
                                                   (short)0, accHi, false, false);
  }
#pragma unroll
  for (int r = 0; r < 8; ++r) {
    int m = hi * 8 + r;
    float inv = (lrow[r] > 0.0f) ? (1.0f / lrow[r]) : 0.0f;
    _Float16* op = o + ((size_t)b * NPIX + q0 + m) * DIM_ + hc0;
    op[lm]      = (_Float16)(accLo[r] * inv);
    op[16 + lm] = (_Float16)(accHi[r] * inv);
  }
}

// ---------------- IRFFN middle: t + dw3x3(t) -> gelu -> bn2, f16 ----------------
__global__ __launch_bounds__(256) void k_ffnmid(const _Float16* __restrict__ t1,
                                                const float* __restrict__ w,
                                                const float* __restrict__ bias,
                                                const float* __restrict__ g,
                                                const float* __restrict__ bb,
                                                const float* __restrict__ mm,
                                                const float* __restrict__ vv,
                                                _Float16* __restrict__ t2) {
  size_t gid = (size_t)blockIdx.x * 256 + threadIdx.x;
  int c = (int)(gid % HID_);
  size_t bn = gid / HID_;
  int b = (int)(bn / NPIX), n = (int)(bn % NPIX);
  int h = n / WS_, wc = n % WS_;
  float acc = bias[c];
#pragma unroll
  for (int dh = 0; dh < 3; ++dh)
#pragma unroll
    for (int dw = 0; dw < 3; ++dw) {
      int hh = h + dh - 1, ww = wc + dw - 1;
      if (hh >= 0 && hh < HS_ && ww >= 0 && ww < WS_)
        acc += (float)t1[((size_t)b * NPIX + hh * WS_ + ww) * HID_ + c] * w[c * 9 + dh * 3 + dw];
    }
  float u = (float)t1[gid] + acc;
  u = gelu_exact(u);
  u = bn_apply(u, g[c], bb[c], mm[c], vv[c]);
  t2[gid] = (_Float16)u;
}

extern "C" void kernel_launch(void* const* d_in, const int* in_sizes, int n_in,
                              void* d_out, int out_size, void* d_ws, size_t ws_size,
                              hipStream_t stream) {
  const float* x      = (const float*)d_in[0];
  const float* lpu_w  = (const float*)d_in[1];
  const float* lpu_b  = (const float*)d_in[2];
  const float* ln1_g  = (const float*)d_in[3];
  const float* ln1_b  = (const float*)d_in[4];
  const float* q_w    = (const float*)d_in[5];
  const float* k_w    = (const float*)d_in[6];
  const float* v_w    = (const float*)d_in[7];
  const float* sr_w   = (const float*)d_in[8];
  const float* sr_b   = (const float*)d_in[9];
  const float* srbn_g = (const float*)d_in[10];
  const float* srbn_b = (const float*)d_in[11];
  const float* srbn_m = (const float*)d_in[12];
  const float* srbn_v = (const float*)d_in[13];
  const float* proj_w = (const float*)d_in[14];
  const float* proj_b = (const float*)d_in[15];
  const float* ln2_g  = (const float*)d_in[16];
  const float* ln2_b  = (const float*)d_in[17];
  const float* fc1_w  = (const float*)d_in[18];
  const float* fc1_b  = (const float*)d_in[19];
  const float* bn1_g  = (const float*)d_in[20];
  const float* bn1_b  = (const float*)d_in[21];
  const float* bn1_m  = (const float*)d_in[22];
  const float* bn1_v  = (const float*)d_in[23];
  const float* dw_w   = (const float*)d_in[24];
  const float* dw_b   = (const float*)d_in[25];
  const float* bn2_g  = (const float*)d_in[26];
  const float* bn2_b  = (const float*)d_in[27];
  const float* bn2_m  = (const float*)d_in[28];
  const float* bn2_v  = (const float*)d_in[29];
  const float* fc2_w  = (const float*)d_in[30];
  const float* fc2_b  = (const float*)d_in[31];
  const float* bn3_g  = (const float*)d_in[32];
  const float* bn3_b  = (const float*)d_in[33];
  const float* bn3_m  = (const float*)d_in[34];
  const float* bn3_v  = (const float*)d_in[35];
  const float* rel    = (const float*)d_in[36];
  float* out = (float*)d_out;

  char* wsp = (char*)d_ws;
  size_t off = 0;
  auto alloc = [&](size_t bytes) -> void* {
    void* p = wsp + off;
    off += (bytes + 255) & ~(size_t)255;
    return p;
  };
  const size_t NBC = (size_t)BN_ * NPIX * DIM_;   // 3,211,264
  const size_t NKC = (size_t)BN_ * NK_ * DIM_;    //   802,816
  const size_t NBH = (size_t)BN_ * NPIX * HID_;   // 12,845,056

  float*    x1   = (float*)   alloc(NBC * 4);
  _Float16* xnh  = (_Float16*)alloc(NBC * 2);
  _Float16* qh   = (_Float16*)alloc(NBC * 2);
  _Float16* xsh  = (_Float16*)alloc(NKC * 2);
  _Float16* kh   = (_Float16*)alloc(NKC * 2);
  _Float16* vTh  = (_Float16*)alloc(NKC * 2);
  _Float16* oh   = (_Float16*)alloc(NBC * 2);
  float*    x2   = (float*)   alloc(NBC * 4);
  _Float16* xn2h = (_Float16*)alloc(NBC * 2);
  _Float16* t1h  = (_Float16*)alloc(NBH * 2);
  _Float16* t2h  = (_Float16*)alloc(NBH * 2);
  _Float16* qwh  = (_Float16*)alloc((size_t)DIM_ * DIM_ * 2);
  _Float16* kwh  = (_Float16*)alloc((size_t)DIM_ * DIM_ * 2);
  _Float16* vwh  = (_Float16*)alloc((size_t)DIM_ * DIM_ * 2);
  _Float16* pwh  = (_Float16*)alloc((size_t)DIM_ * DIM_ * 2);
  _Float16* f1wh = (_Float16*)alloc((size_t)HID_ * DIM_ * 2);
  _Float16* f2wh = (_Float16*)alloc((size_t)DIM_ * HID_ * 2);

  auto cvt = [&](const float* s, _Float16* d, int n) {
    k_cvt<<<(n + 255) / 256, 256, 0, stream>>>(s, d, n);
  };
  cvt(q_w, qwh, DIM_ * DIM_);
  cvt(k_w, kwh, DIM_ * DIM_);
  cvt(v_w, vwh, DIM_ * DIM_);
  cvt(proj_w, pwh, DIM_ * DIM_);
  cvt(fc1_w, f1wh, HID_ * DIM_);
  cvt(fc2_w, f2wh, DIM_ * HID_);

  const int Mq = BN_ * NPIX;   // 12544
  const int Mk = BN_ * NK_;    // 3136

  // 1) LPU
  k_lpu<<<Mq, 256, 0, stream>>>(x, lpu_w, lpu_b, x1);
  // 2) LN1 -> f16
  k_ln<<<Mq, 256, 0, stream>>>(x1, ln1_g, ln1_b, xnh);
  // 3) q = xn @ q_w^T
  k_gemm<<<dim3(Mq / 16, DIM_ / 128), 256, 0, stream>>>(
      xnh, qwh, Mq, DIM_, DIM_, 0,
      nullptr, nullptr, nullptr, nullptr, nullptr, nullptr, nullptr, qh, 0);
  // 4) spatial reduction + BN
  k_sr<<<Mk, 256, 0, stream>>>(xnh, sr_w, sr_b, srbn_g, srbn_b, srbn_m, srbn_v, xsh);
  // 5) k, v (v stored transposed per batch: [C, Nk])
  k_gemm<<<dim3(Mk / 16, DIM_ / 128), 256, 0, stream>>>(
      xsh, kwh, Mk, DIM_, DIM_, 0,
      nullptr, nullptr, nullptr, nullptr, nullptr, nullptr, nullptr, kh, 0);
  k_gemm<<<dim3(Mk / 16, DIM_ / 128), 256, 0, stream>>>(
      xsh, vwh, Mk, DIM_, DIM_, 1,
      nullptr, nullptr, nullptr, nullptr, nullptr, nullptr, nullptr, vTh, NK_);
  // 6) fused attention
  k_attn<<<(BN_ * HEADS_ * (NPIX / 16)) / 8, 256, 0, stream>>>(qh, kh, vTh, rel, oh);
  // 7) proj + bias + residual(x1) -> x2 (f32)
  k_gemm<<<dim3(Mq / 16, DIM_ / 128), 256, 0, stream>>>(
      oh, pwh, Mq, DIM_, DIM_, 2,
      proj_b, nullptr, nullptr, nullptr, nullptr, x1, x2, nullptr, 0);
  // 8) LN2 -> f16
  k_ln<<<Mq, 256, 0, stream>>>(x2, ln2_g, ln2_b, xn2h);
  // 9) fc1 + gelu + bn1 -> t1 (f16)
  k_gemm<<<dim3(Mq / 16, HID_ / 128), 256, 0, stream>>>(
      xn2h, f1wh, Mq, HID_, DIM_, 3,
      fc1_b, bn1_g, bn1_b, bn1_m, bn1_v, nullptr, nullptr, t1h, 0);
  // 10) t + dw3x3(t) -> gelu -> bn2 -> t2 (f16)
  k_ffnmid<<<(int)(NBH / 256), 256, 0, stream>>>(t1h, dw_w, dw_b,
                                                 bn2_g, bn2_b, bn2_m, bn2_v, t2h);
  // 11) fc2 + bias + bn3 + residual(x2) -> out (f32)
  k_gemm<<<dim3(Mq / 16, DIM_ / 128), 256, 0, stream>>>(
      t2h, f2wh, Mq, DIM_, HID_, 4,
      fc2_b, bn3_g, bn3_b, bn3_m, bn3_v, x2, out, nullptr, 0);
  (void)in_sizes; (void)n_in; (void)out_size; (void)ws_size;
}
